// Head_10453950398691
// MI455X (gfx1250) — matrix operand order repaired and checked by reading, not compiled, not run on hardware
//
#include <hip/hip_runtime.h>
#include <hip/hip_bf16.h>

// ---------------- problem constants ----------------
#define N_EMBD   768
#define HEAD     72
#define HEADP    96      // head dim padded to 6x16 WMMA tiles
#define SEQ      512
#define BATCH    64
#define MROWS    (BATCH*SEQ)        // 32768
#define NCH      (N_EMBD/32)        // 24 k-chunks
#define QK_SCALE 0.11785113019775793f   // 1/sqrt(72)

typedef __attribute__((ext_vector_type(16))) _Float16 v16h;
typedef __attribute__((ext_vector_type(8)))  float    v8f;

union V16HU { uint4 u[2]; v16h h; };

__device__ __forceinline__ v16h make_v16h(uint4 lo, uint4 hi) {
    V16HU c; c.u[0] = lo; c.u[1] = hi; return c.h;
}

// =====================================================================
// Kernel 0: one-shot weight prep.  W[768][72] f32 (q/k/v) ->
// Wt[z][96][768] f16, transposed (n-major) + zero-padded n=72..95,
// with 1/sqrt(72) folded into Wq.  Tiny (663 KB read), runs once.
// =====================================================================
__global__ __launch_bounds__(256) void prep_weights_kernel(
    const float* __restrict__ Wk,
    const float* __restrict__ Wq,
    const float* __restrict__ Wv,
    _Float16* __restrict__ Wt)
{
    const int idx = blockIdx.x * 256 + threadIdx.x;
    if (idx >= 3 * HEADP * N_EMBD) return;
    const int z   = idx / (HEADP * N_EMBD);
    const int rem = idx % (HEADP * N_EMBD);
    const int n   = rem / N_EMBD;
    const int k   = rem % N_EMBD;
    const float* W = (z == 0) ? Wq : (z == 1) ? Wk : Wv;
    const float sc = (z == 0) ? QK_SCALE : 1.0f;
    const float v  = (n < HEAD) ? W[k * HEAD + n] * sc : 0.0f;
    Wt[idx] = (_Float16)v;
}

// ---------------- projection helpers ----------------
__device__ __forceinline__ void load_b6(const _Float16* __restrict__ Wz, int c,
                                        int lane15, int bkb, uint4 b[6][2]) {
    #pragma unroll
    for (int t = 0; t < 6; ++t) {
        const uint4* bp = (const uint4*)(Wz + (size_t)(16 * t + lane15) * N_EMBD + c * 32 + bkb);
        b[t][0] = bp[0];
        b[t][1] = bp[2];
    }
}

__device__ __forceinline__ void load_a4(const float* __restrict__ xp, float4 a[4]) {
    a[0] = ((const float4*)xp)[0];
    a[1] = ((const float4*)xp)[1];
    a[2] = ((const float4*)(xp + 16))[0];
    a[3] = ((const float4*)(xp + 16))[1];
}

__device__ __forceinline__ v16h cvt_a(const float4 f[4]) {
    v16h a;
    a[0]=(_Float16)f[0].x;  a[1]=(_Float16)f[0].y;  a[2]=(_Float16)f[0].z;  a[3]=(_Float16)f[0].w;
    a[4]=(_Float16)f[1].x;  a[5]=(_Float16)f[1].y;  a[6]=(_Float16)f[1].z;  a[7]=(_Float16)f[1].w;
    a[8]=(_Float16)f[2].x;  a[9]=(_Float16)f[2].y;  a[10]=(_Float16)f[2].z; a[11]=(_Float16)f[2].w;
    a[12]=(_Float16)f[3].x; a[13]=(_Float16)f[3].y; a[14]=(_Float16)f[3].z; a[15]=(_Float16)f[3].w;
    return a;
}

// =====================================================================
// Kernel 1: QKV projection.  grid = (MROWS/128, 3), block = 128 (4 waves)
// Each wave owns a 32-row M tile (two A tiles sharing the B tiles),
// software-pipelined over k-chunks with explicit double buffering.
//   z==0 -> Q (pre-scaled), row-major f16 [MROWS][96]
//   z==1 -> K,              row-major f16 [MROWS][96]
//   z==2 -> V, transposed f16 [BATCH][96][512]
// =====================================================================
__global__ __launch_bounds__(128) void proj_qkv_kernel(
    const float* __restrict__ x,
    const _Float16* __restrict__ Wt,
    _Float16* __restrict__ Qh,
    _Float16* __restrict__ Kh,
    _Float16* __restrict__ Vt)
{
    const int tid    = threadIdx.x;
    const int lane   = tid & 31;
    const int wv     = tid >> 5;
    const int z      = blockIdx.y;
    const int lane15 = lane & 15;
    const int hi8    = (lane >= 16) ? 8 : 0;
    const int akb    = (lane < 16) ? 0 : 8;     // A k-half offset
    const int bkb    = (lane < 16) ? 0 : 8;     // B k-half offset

    const _Float16* Wz = Wt + (size_t)z * HEADP * N_EMBD;

    const int m0 = blockIdx.x * 128 + wv * 32;  // first row of this wave's 32-row tile
    const float* x0 = x + (size_t)(m0 + lane15) * N_EMBD + akb;
    const float* x1 = x + (size_t)(m0 + 16 + lane15) * N_EMBD + akb;

    v8f acc0[6] = {}, acc1[6] = {};

    // ---- double-buffered pipeline: P = even-chunk bufs, Nx = odd ----
    uint4  bP[6][2], bN[6][2];
    float4 aP0[4], aP1[4], aN0[4], aN1[4];

    load_b6(Wz, 0, lane15, bkb, bP);
    load_a4(x0, aP0);
    load_a4(x1, aP1);

    for (int c = 0; c < NCH; c += 2) {
        // issue odd-chunk loads, then consume even chunk
        load_b6(Wz, c + 1, lane15, bkb, bN);
        load_a4(x0 + (c + 1) * 32, aN0);
        load_a4(x1 + (c + 1) * 32, aN1);
        {
            const v16h a0 = cvt_a(aP0);
            const v16h a1 = cvt_a(aP1);
            #pragma unroll
            for (int t = 0; t < 6; ++t) {
                const v16h bm = make_v16h(bP[t][0], bP[t][1]);
                acc0[t] = __builtin_amdgcn_wmma_f32_16x16x32_f16(false, a0, false, bm, (short)0, acc0[t], false, false);
                acc1[t] = __builtin_amdgcn_wmma_f32_16x16x32_f16(false, a1, false, bm, (short)0, acc1[t], false, false);
            }
        }
        // issue next even-chunk loads, then consume odd chunk
        if (c + 2 < NCH) {
            load_b6(Wz, c + 2, lane15, bkb, bP);
            load_a4(x0 + (c + 2) * 32, aP0);
            load_a4(x1 + (c + 2) * 32, aP1);
        }
        {
            const v16h a0 = cvt_a(aN0);
            const v16h a1 = cvt_a(aN1);
            #pragma unroll
            for (int t = 0; t < 6; ++t) {
                const v16h bm = make_v16h(bN[t][0], bN[t][1]);
                acc0[t] = __builtin_amdgcn_wmma_f32_16x16x32_f16(false, a0, false, bm, (short)0, acc0[t], false, false);
                acc1[t] = __builtin_amdgcn_wmma_f32_16x16x32_f16(false, a1, false, bm, (short)0, acc1[t], false, false);
            }
        }
    }

    // ---- store ----
    if (z < 2) {
        _Float16* out = (z == 0) ? Qh : Kh;
        #pragma unroll
        for (int t = 0; t < 6; ++t) {
            const int h = 16 * t + lane15;
            #pragma unroll
            for (int r = 0; r < 8; ++r) {
                out[(size_t)(m0 + r + hi8) * HEADP + h]      = (_Float16)acc0[t][r];
                out[(size_t)(m0 + 16 + r + hi8) * HEADP + h] = (_Float16)acc1[t][r];
            }
        }
    } else {
        // V transposed: Vt[b][h][t_seq]; pack 8 consecutive seq positions -> b128 store
        const int b   = m0 / SEQ;
        const int ts0 = (m0 % SEQ) + hi8;
        #pragma unroll
        for (int t = 0; t < 6; ++t) {
            const int h = 16 * t + lane15;
            union { uint4 u; _Float16 f[8]; } p0, p1;
            #pragma unroll
            for (int r = 0; r < 8; ++r) { p0.f[r] = (_Float16)acc0[t][r]; p1.f[r] = (_Float16)acc1[t][r]; }
            *(uint4*)&Vt[((size_t)(b * HEADP + h)) * SEQ + ts0]      = p0.u;
            *(uint4*)&Vt[((size_t)(b * HEADP + h)) * SEQ + ts0 + 16] = p1.u;
        }
    }
}

// =====================================================================
// Kernel 2: causal flash attention. grid = (SEQ/64, BATCH), block = 128.
// Each wave independently handles one 16-query tile (online softmax,
// 32-key blocks).  All 24 K/V b128 loads of a key block are issued in
// one batch before the first WMMA; P is re-striped C->A layout via a
// per-wave LDS tile guarded by s_wait_dscnt.
// =====================================================================
__global__ __launch_bounds__(128) void flash_attn_kernel(
    const _Float16* __restrict__ Qh,
    const _Float16* __restrict__ Kh,
    const _Float16* __restrict__ Vt,
    float* __restrict__ out)
{
    __shared__ __align__(16) _Float16 Plds[4 * 16 * 32];   // 1 KB per wave

    const int tid    = threadIdx.x;
    const int lane   = tid & 31;
    const int wv     = tid >> 5;
    const int lane15 = lane & 15;
    const int hi8    = (lane >= 16) ? 8 : 0;
    const int akb    = (lane < 16) ? 0 : 8;     // A k-half offset
    const int bkb    = (lane < 16) ? 0 : 8;     // B k-half offset

    const int b  = blockIdx.y;
    const int qt = blockIdx.x * 64 + wv * 16;   // first query row of this wave

    const _Float16* Kb = Kh + (size_t)b * SEQ * HEADP;
    const _Float16* Vb = Vt + (size_t)b * HEADP * SEQ;
    _Float16*       pw = &Plds[wv * 16 * 32];

    // ---- load Q A-operands once (3 head-dim chunks, pre-scaled) ----
    const _Float16* qrow = Qh + (size_t)(b * SEQ + qt + lane15) * HEADP;
    v16h qa[3];
    #pragma unroll
    for (int c = 0; c < 3; ++c) {
        const uint4 lo = *(const uint4*)(qrow + c * 32 + akb);
        const uint4 hi = *(const uint4*)(qrow + c * 32 + akb + 16);
        qa[c] = make_v16h(lo, hi);
    }

    v8f o[6] = {};
    float Mr[8], Lr[8];
    #pragma unroll
    for (int r = 0; r < 8; ++r) { Mr[r] = -__builtin_inff(); Lr[r] = 0.0f; }

    const int nkb = (qt + 16 + 31) >> 5;        // causal key-block count
    for (int kb = 0; kb < nkb; ++kb) {
        // ---- issue ALL global loads for this key block in one batch ----
        uint4 kr[6][2];                          // [c*2+s] -> K B-operand
        #pragma unroll
        for (int c = 0; c < 3; ++c) {
            #pragma unroll
            for (int s = 0; s < 2; ++s) {
                const uint4* kp = (const uint4*)(Kb + (size_t)(kb * 32 + s * 16 + lane15) * HEADP + c * 32 + bkb);
                kr[c * 2 + s][0] = kp[0];
                kr[c * 2 + s][1] = kp[2];
            }
        }
        uint4 vr[6][2];
        #pragma unroll
        for (int t = 0; t < 6; ++t) {
            const uint4* vp = (const uint4*)(Vb + (size_t)(16 * t + lane15) * SEQ + kb * 32 + bkb);
            vr[t][0] = vp[0];
            vr[t][1] = vp[2];
        }

        // ---- prefetch next key block while this one computes ----
        if (kb + 1 < nkb) {
            __builtin_prefetch(Kb + (size_t)((kb + 1) * 32 + lane15) * HEADP, 0, 0);
            __builtin_prefetch(Vb + (size_t)lane15 * SEQ + (kb + 1) * 32, 0, 0);
        }

        // ---- S = Q * K^T for two 16-key sub-tiles ----
        v8f s0 = {}, s1 = {};
        #pragma unroll
        for (int c = 0; c < 3; ++c) {
            s0 = __builtin_amdgcn_wmma_f32_16x16x32_f16(false, qa[c], false,
                    make_v16h(kr[c * 2][0], kr[c * 2][1]), (short)0, s0, false, false);
            s1 = __builtin_amdgcn_wmma_f32_16x16x32_f16(false, qa[c], false,
                    make_v16h(kr[c * 2 + 1][0], kr[c * 2 + 1][1]), (short)0, s1, false, false);
        }

        // ---- causal mask + online softmax (row m = r + hi8) ----
        const int nk0 = kb * 32 + lane15;
        const int nk1 = nk0 + 16;
        #pragma unroll
        for (int r = 0; r < 8; ++r) {
            const int mq = qt + r + hi8;
            const float v0 = (nk0 <= mq) ? s0[r] : -__builtin_inff();
            const float v1 = (nk1 <= mq) ? s1[r] : -__builtin_inff();
            float mx = fmaxf(v0, v1);
            #pragma unroll
            for (int off = 1; off < 16; off <<= 1)
                mx = fmaxf(mx, __shfl_xor(mx, off, 32));
            const float newM  = fmaxf(Mr[r], mx);
            const float scale = __expf(Mr[r] - newM);
            Mr[r] = newM;
            const float p0 = __expf(v0 - newM);
            const float p1 = __expf(v1 - newM);
            float rs = p0 + p1;
            #pragma unroll
            for (int off = 1; off < 16; off <<= 1)
                rs += __shfl_xor(rs, off, 32);
            Lr[r] = Lr[r] * scale + rs;
            #pragma unroll
            for (int t = 0; t < 6; ++t) o[t][r] *= scale;
            // P back into per-wave LDS tile [16][32] (C-layout positions)
            pw[(r + hi8) * 32 + lane15]      = (_Float16)p0;
            pw[(r + hi8) * 32 + 16 + lane15] = (_Float16)p1;
        }

        // same-wave DS RAW: wait for the 16 ds_stores before re-striping
        asm volatile("s_wait_dscnt 0x0" ::: "memory");

        // ---- P as A-operand (16x32 over keys) ----
        const uint4 plo = *(const uint4*)(pw + lane15 * 32 + akb);
        const uint4 phi = *(const uint4*)(pw + lane15 * 32 + akb + 16);
        const v16h pa = make_v16h(plo, phi);

        // ---- O += P * V ----
        #pragma unroll
        for (int t = 0; t < 6; ++t) {
            o[t] = __builtin_amdgcn_wmma_f32_16x16x32_f16(
                false, pa, false, make_v16h(vr[t][0], vr[t][1]), (short)0, o[t], false, false);
        }
    }

    // ---- normalize and store (only real head dims 0..71) ----
    float invL[8];
    #pragma unroll
    for (int r = 0; r < 8; ++r) invL[r] = 1.0f / Lr[r];
    #pragma unroll
    for (int t = 0; t < 5; ++t) {
        const int h = 16 * t + lane15;
        if (h < HEAD) {
            #pragma unroll
            for (int r = 0; r < 8; ++r) {
                out[(size_t)(b * SEQ + qt + r + hi8) * HEAD + h] = o[t][r] * invL[r];
            }
        }
    }
}

// =====================================================================
extern "C" void kernel_launch(void* const* d_in, const int* in_sizes, int n_in,
                              void* d_out, int out_size, void* d_ws, size_t ws_size,
                              hipStream_t stream) {
    const float* x  = (const float*)d_in[0];
    const float* Wk = (const float*)d_in[1];
    const float* Wq = (const float*)d_in[2];
    const float* Wv = (const float*)d_in[3];
    float* out = (float*)d_out;

    // workspace: Qh | Kh ([MROWS][96] f16) | Vt ([B][96][512] f16) | Wt ([3][96][768] f16)
    _Float16* Qh = (_Float16*)d_ws;
    _Float16* Kh = Qh + (size_t)MROWS * HEADP;
    _Float16* Vt = Kh + (size_t)MROWS * HEADP;
    _Float16* Wt = Vt + (size_t)BATCH * HEADP * SEQ;

    const int wtot = 3 * HEADP * N_EMBD;
    prep_weights_kernel<<<(wtot + 255) / 256, 256, 0, stream>>>(Wk, Wq, Wv, Wt);
    proj_qkv_kernel<<<dim3(MROWS / 128, 3), 128, 0, stream>>>(x, Wt, Qh, Kh, Vt);
    flash_attn_kernel<<<dim3(SEQ / 64, BATCH), 128, 0, stream>>>(Qh, Kh, Vt, out);
}
